// SGLFourOverSixWeightOnlyLinear_39109972197857
// MI455X (gfx1250) — matrix operand order repaired and checked by reading, not compiled, not run on hardware
//
#include <hip/hip_runtime.h>

typedef __attribute__((ext_vector_type(16))) _Float16 v16h;
typedef __attribute__((ext_vector_type(2)))  _Float16 v2h;
typedef __attribute__((ext_vector_type(8)))  float    v8f;
typedef __attribute__((ext_vector_type(4)))  float    v4f;
typedef __attribute__((ext_vector_type(4)))  int      v4i;
typedef __attribute__((ext_vector_type(4)))  unsigned v4u;
typedef __attribute__((ext_vector_type(8)))  unsigned v8u;

// FP4 E2M1 -> f16 pair, branchless: one v_perm_b32 byte LUT + shift/mask for
// signs, one v_and_or_b32 merge, one v_pk_mul_f16 by (block_scale * amax).
static __device__ inline unsigned fp4_pair_to_h2(unsigned w, v2h hs2) {
    unsigned sel  = ((w << 8) | (w << 20)) & 0x07000700u;   // codes -> sel bytes 1,3
    unsigned mag  = __builtin_amdgcn_perm(0x46444240u, 0x3E3C3800u, sel);
    unsigned sgn  = (w << 12) | (w << 24);                  // bit3->15, bit7->31
    unsigned bits = (sgn & 0x80008000u) | mag;              // v_and_or_b32
    v2h p = __builtin_bit_cast(v2h, bits) * hs2;
    return __builtin_bit_cast(unsigned, p);
}

// Pre-pass: convert fp32 activations to f16 once (row-major [M,K]).
__global__ __launch_bounds__(256) void a_to_half_kernel(
    const float* __restrict__ in, _Float16* __restrict__ out, int n4) {
    int i = blockIdx.x * 256 + threadIdx.x;
    if (i < n4) {
        v4f f = ((const v4f*)in)[i];
        v2h lo, hi;
        lo[0] = (_Float16)f[0]; lo[1] = (_Float16)f[1];
        hi[0] = (_Float16)f[2]; hi[1] = (_Float16)f[3];
        ((v2h*)out)[2 * i]     = lo;
        ((v2h*)out)[2 * i + 1] = hi;
    }
}

// Split-K reduction: out[m][n] = sum_kc partial[kc][m][n] + bias[n].
// Fixed summation order -> deterministic.
__global__ __launch_bounds__(256) void reduce_bias_kernel(
    const float* __restrict__ part, const float* __restrict__ bias,
    float* __restrict__ out, int total4, int N4, int ks, int strideMN4) {
    int i = blockIdx.x * 256 + threadIdx.x;
    if (i < total4) {
        v4f s = ((const v4f*)part)[i];
        for (int kc = 1; kc < ks; ++kc)
            s += ((const v4f*)part)[i + (long)kc * strideMN4];
        s += ((const v4f*)bias)[i % N4];
        ((v4f*)out)[i] = s;
    }
}

// One block = 8 waves; each wave computes a 16-wide N strip for all 32 M rows
// over K range [blockIdx.y*Kchunk, +Kchunk). Two 16x16 f32 accumulators share
// one B fragment per step. Requires Kchunk % 64 == 0 and Kchunk >= 64.
template <bool AHALF>
__global__ __launch_bounds__(256) void fp4_linear_wmma_kernel(
    const void*  __restrict__ aptr,     // [M=32, K] f32 or f16 (AHALF)
    const int*   __restrict__ qweight,  // [N, K/2] int32, 1 byte (2 nibbles) each
    const float* __restrict__ scales,   // [N, K/16] f32
    const float* __restrict__ amax,     // [1]
    const float* __restrict__ bias,     // [N]
    float*       __restrict__ dst,      // [M,N] out  or  [KS][M,N] partials
    int K, int N, int Kchunk, int addBias)
{
    const int lane = threadIdx.x & 31;
    const int wave = threadIdx.x >> 5;
    const int n0   = blockIdx.x * 128 + wave * 16;
    const int col  = lane & 15;
    const int hi   = lane >> 4;          // 0 or 1 (half-wave)
    const int n    = n0 + col;

    const float am = amax[0];
    const int Kq = K >> 1;               // int32 per qweight row
    const int Ks = K >> 4;               // scales per row
    const int kBegin = blockIdx.y * Kchunk;
    float* outb = dst + (long)blockIdx.y * 32 * N;

    // pointer-increment induction (keeps per-iteration address math to adds)
    const char*     qp  = (const char*)(qweight + (long)n * Kq)
                          + (long)((kBegin + hi * 16) >> 1) * 4;
    const float*    sp  = scales + (long)n * Ks + ((kBegin + hi * 16) >> 4);
    const float*    fp0 = (const float*)aptr    + (long)col * K        + kBegin + hi * 16;
    const float*    fp1 = (const float*)aptr    + (long)(col + 16) * K + kBegin + hi * 16;
    const _Float16* hp0 = (const _Float16*)aptr + (long)col * K        + kBegin + hi * 16;
    const _Float16* hp1 = (const _Float16*)aptr + (long)(col + 16) * K + kBegin + hi * 16;

    v8f c0 = {};
    v8f c1 = {};

    int it = Kchunk >> 6;                // two 32-K steps per iteration, >= 1
    do {
        #pragma unroll
        for (int u = 0; u < 2; ++u) {
            // -------- B fragment: lane = column n, 16 K-values ----------
            v4i q0 = *(const v4i*)(qp + 64 * u);
            v4i q1 = *(const v4i*)(qp + 64 * u + 16);

            const _Float16 hs = (_Float16)(sp[2 * u] * am);
            v2h hs2; hs2[0] = hs; hs2[1] = hs;

            v8u bw;
            #pragma unroll
            for (int j = 0; j < 4; ++j) {
                bw[j]     = fp4_pair_to_h2((unsigned)q0[j], hs2);
                bw[4 + j] = fp4_pair_to_h2((unsigned)q1[j], hs2);
            }
            v16h b = __builtin_bit_cast(v16h, bw);

            // -------- A fragments ---------------------------------------
            v16h a0, a1;
            if constexpr (AHALF) {
                v4u x0 = *(const v4u*)(hp0 + 32 * u);
                v4u x1 = *(const v4u*)(hp0 + 32 * u + 8);
                v4u y0 = *(const v4u*)(hp1 + 32 * u);
                v4u y1 = *(const v4u*)(hp1 + 32 * u + 8);
                v8u ua, ub;
                #pragma unroll
                for (int i = 0; i < 4; ++i) {
                    ua[i] = x0[i]; ua[4 + i] = x1[i];
                    ub[i] = y0[i]; ub[4 + i] = y1[i];
                }
                a0 = __builtin_bit_cast(v16h, ua);
                a1 = __builtin_bit_cast(v16h, ub);
            } else {
                v4f f0 = *(const v4f*)(fp0 + 32 * u);
                v4f f1 = *(const v4f*)(fp0 + 32 * u + 4);
                v4f f2 = *(const v4f*)(fp0 + 32 * u + 8);
                v4f f3 = *(const v4f*)(fp0 + 32 * u + 12);
                v4f g0 = *(const v4f*)(fp1 + 32 * u);
                v4f g1 = *(const v4f*)(fp1 + 32 * u + 4);
                v4f g2 = *(const v4f*)(fp1 + 32 * u + 8);
                v4f g3 = *(const v4f*)(fp1 + 32 * u + 12);
                #pragma unroll
                for (int i = 0; i < 4; ++i) {
                    a0[i]    = (_Float16)f0[i];  a0[4+i]  = (_Float16)f1[i];
                    a0[8+i]  = (_Float16)f2[i];  a0[12+i] = (_Float16)f3[i];
                    a1[i]    = (_Float16)g0[i];  a1[4+i]  = (_Float16)g1[i];
                    a1[8+i]  = (_Float16)g2[i];  a1[12+i] = (_Float16)g3[i];
                }
            }

            // D = A x B + C  (v_wmma_f32_16x16x32_f16)
            c0 = __builtin_amdgcn_wmma_f32_16x16x32_f16(
                    false, a0, false, b, (short)0, c0, false, false);
            c1 = __builtin_amdgcn_wmma_f32_16x16x32_f16(
                    false, a1, false, b, (short)0, c1, false, false);
        }

        // speculative prefetch ~2 iterations ahead of the weight stream
        __builtin_prefetch((const void*)(qp + 256), 0, 1);

        qp  += 128;                      // 2 steps * 16 int32
        sp  += 4;                        // 2 steps * 2 blocks
        fp0 += 64; fp1 += 64;            // 2 steps * 32 K
        hp0 += 64; hp1 += 64;
    } while (--it > 0);

    // ---------- epilogue: C VGPR r <-> M = hi*8 + r ; lane col = N column ----
    const float bv = addBias ? bias[n] : 0.0f;
    const int   mb = hi * 8;
    #pragma unroll
    for (int r = 0; r < 8; ++r) {
        outb[(long)(mb + r)      * N + n] = c0[r] + bv;
        outb[(long)(mb + r + 16) * N + n] = c1[r] + bv;
    }
}

extern "C" void kernel_launch(void* const* d_in, const int* in_sizes, int n_in,
                              void* d_out, int out_size, void* d_ws, size_t ws_size,
                              hipStream_t stream) {
    const float* inp     = (const float*)d_in[0];
    const int*   qweight = (const int*)  d_in[1];
    const float* scales  = (const float*)d_in[2];
    const float* amax    = (const float*)d_in[3];
    const float* bias    = (const float*)d_in[4];
    float*       out     = (float*)d_out;

    const int  N = in_sizes[4];                       // bias length = out_features
    const long K = (2L * (long)in_sizes[1]) / N;      // qweight = N * K/2
    const long M = (long)in_sizes[0] / K;             // == 32
    const long aElems = M * K;

    const size_t aBytes   = (size_t)aElems * sizeof(_Float16);
    const size_t aAligned = (aBytes + 255) & ~(size_t)255;
    const int    KS       = 4;                        // split-K factor
    const size_t partBytes = (size_t)KS * M * N * sizeof(float);

    dim3 block(256, 1, 1);

    if (ws_size >= aAligned + partBytes && (K / KS) % 64 == 0) {
        // full path: f16 A staging + split-K x4 for occupancy
        _Float16* ah   = (_Float16*)d_ws;
        float*    part = (float*)((char*)d_ws + aAligned);
        const int n4 = (int)(aElems / 4);
        a_to_half_kernel<<<dim3((n4 + 255) / 256), block, 0, stream>>>(inp, ah, n4);

        dim3 grid((unsigned)(N / 128), KS, 1);
        fp4_linear_wmma_kernel<true><<<grid, block, 0, stream>>>(
            (const void*)ah, qweight, scales, amax, bias, part,
            (int)K, N, (int)(K / KS), /*addBias=*/0);

        const int total4 = (int)(M * N / 4);
        reduce_bias_kernel<<<dim3((total4 + 255) / 256), block, 0, stream>>>(
            part, bias, out, total4, N / 4, KS, (int)(M * N / 4));
    } else if (ws_size >= aBytes) {
        _Float16* ah = (_Float16*)d_ws;
        const int n4 = (int)(aElems / 4);
        a_to_half_kernel<<<dim3((n4 + 255) / 256), block, 0, stream>>>(inp, ah, n4);
        dim3 grid((unsigned)(N / 128), 1, 1);
        fp4_linear_wmma_kernel<true><<<grid, block, 0, stream>>>(
            (const void*)ah, qweight, scales, amax, bias, out,
            (int)K, N, (int)K, /*addBias=*/1);
    } else {
        dim3 grid((unsigned)(N / 128), 1, 1);
        fp4_linear_wmma_kernel<false><<<grid, block, 0, stream>>>(
            (const void*)inp, qweight, scales, amax, bias, out,
            (int)K, N, (int)K, /*addBias=*/1);
    }
}
